// NodeModel_21827023798511
// MI455X (gfx1250) — matrix operand order repaired and checked by reading, not compiled, not run on hardware
//
#include <hip/hip_runtime.h>
#include <hip/hip_bf16.h>

typedef unsigned short u16;
typedef __attribute__((ext_vector_type(8)))  u16    u16x8;
typedef __attribute__((ext_vector_type(16))) __bf16 v16bf;
typedef __attribute__((ext_vector_type(8)))  float  v8f;

#define N_NODES_C  50000
#define N_EDGES_C  800000
#define N_GRAPHS_C 64
#define EDGE_F_C   512
#define NODE_F_C   9
#define HID_C      512
#define U_F_C      256

#define K1_ORIG 521   // NODE_F + EDGE_F
#define K1_PAD  544   // 17 * 32
#define K2_PAD  512   // 16 * 32
#define K3_ORIG 777   // NODE_F + HID + U_F
#define K3_PAD  800   // 25 * 32

#define EDGE_M  64    // edges per workgroup (4 M-tiles per wave)
#define NODE_M  64    // nodes per workgroup

union Frag { v16bf v; u16x8 h[2]; };

__device__ __forceinline__ u16 f2bf(float f) {
    union { float f; unsigned u; } c; c.f = f;
    unsigned u = c.u;
    u += 0x7FFFu + ((u >> 16) & 1u);      // round to nearest even
    return (u16)(u >> 16);
}

// ---------------------------------------------------------------------------
// Pack a row-major f32 weight [Korig x 512] into bf16 WMMA-B-fragment layout,
// K zero-padded to Kpad. Packed flat index:
//   ((kt*32 + nt)*32 + lane)*16 + i   with k = kt*32 + (i/8)*16 + (lane/16)*8 + i%8,
//                                         n = nt*16 + (lane%16)
// so each lane's 16 bf16 B-fragment values are contiguous (two b128 loads).
// ---------------------------------------------------------------------------
__global__ void pack_w_kernel(const float* __restrict__ W, u16* __restrict__ Wp,
                              int Korig, int Kpad) {
    int tid = blockIdx.x * blockDim.x + threadIdx.x;
    int total = Kpad * 512;
    if (tid >= total) return;
    int i    = tid & 15;
    int lane = (tid >> 4) & 31;
    int nt   = (tid >> 9) & 31;
    int kt   = tid >> 14;
    int k = kt * 32 + (i >> 3) * 16 + (lane >> 4) * 8 + (i & 7);
    int n = nt * 16 + (lane & 15);
    float v = (k < Korig) ? W[(long)k * 512 + n] : 0.0f;
    Wp[tid] = f2bf(v);
}

// ---------------------------------------------------------------------------
// Per-destination-node edge counts (for scatter_mean).
// ---------------------------------------------------------------------------
__global__ void count_kernel(const int* __restrict__ ei, float* __restrict__ cnt) {
    long e = (long)blockIdx.x * blockDim.x + threadIdx.x;
    if (e < N_EDGES_C) atomicAdd(&cnt[ei[e]], 1.0f);
}

// ---------------------------------------------------------------------------
// Fused edge MLP: h = relu(relu([x[col],ea] @ W1 + b1) @ W2 + b2),
// then atomic scatter-add into aggSum[row].
// WG = 256 threads (8 waves), M = 64 edges, each wave owns 64 columns.
// LDS: 64KB h-buffer + 4KB A-tile + indices (~70KB; gfx1250 WGP LDS = 320KB).
// ---------------------------------------------------------------------------
__global__ __launch_bounds__(256) void edge_mlp_kernel(
    const float* __restrict__ x, const int* __restrict__ ei,
    const float* __restrict__ ea, const float* __restrict__ b1,
    const float* __restrict__ b2, const u16* __restrict__ W1p,
    const u16* __restrict__ W2p, float* __restrict__ aggSum) {

    __shared__ __align__(16) u16 hbuf[EDGE_M * 512];      // 64 KB
    __shared__ __align__(16) u16 atile[EDGE_M * 32];      // 4 KB
    __shared__ int rowL[EDGE_M];
    __shared__ int colL[EDGE_M];

    const int  t      = threadIdx.x;
    const int  wave   = t >> 5;
    const int  lane   = t & 31;
    const int  laneHi = lane >> 4;   // 0/1
    const int  laneLo = lane & 15;
    const long eBase  = (long)blockIdx.x * EDGE_M;

    if (t < EDGE_M) {
        long e  = eBase + t;
        rowL[t] = (e < N_EDGES_C) ? ei[e] : 0;
        colL[t] = (e < N_EDGES_C) ? ei[N_EDGES_C + e] : 0;
    }
    __syncthreads();

    const v8f zero8 = {0.f, 0.f, 0.f, 0.f, 0.f, 0.f, 0.f, 0.f};
    v8f acc[4][4];
    #pragma unroll
    for (int mt = 0; mt < 4; ++mt)
        #pragma unroll
        for (int nt = 0; nt < 4; ++nt) acc[mt][nt] = zero8;

    // ---------------- Phase 1: [E,544] @ W1 -> h (LDS) ----------------
    for (int kt = 0; kt < K1_PAD / 32; ++kt) {
        __syncthreads();
        #pragma unroll
        for (int q = 0; q < (EDGE_M * 32) / 256; ++q) {
            int idx = q * 256 + t;
            int r = idx >> 5, c = idx & 31;
            int k = kt * 32 + c;
            long e = eBase + r;
            float v = 0.0f;
            if (e < N_EDGES_C) {
                if (k < NODE_F_C)      v = x[(long)colL[r] * NODE_F_C + k];
                else if (k < K1_ORIG)  v = ea[e * (long)EDGE_F_C + (k - NODE_F_C)];
            }
            atile[idx] = f2bf(v);
        }
        __syncthreads();

        Frag a[4];
        #pragma unroll
        for (int mt = 0; mt < 4; ++mt) {
            const u16* p = &atile[(mt * 16 + laneLo) * 32 + laneHi * 8];
            a[mt].h[0] = *(const u16x8*)p;
            a[mt].h[1] = *(const u16x8*)(p + 16);
        }
        #pragma unroll
        for (int nt = 0; nt < 4; ++nt) {
            int ntg = wave * 4 + nt;
            const u16* bp = &W1p[(((long)kt * 32 + ntg) * 32 + lane) * 16];
            Frag b;
            b.h[0] = *(const u16x8*)bp;
            b.h[1] = *(const u16x8*)(bp + 8);
            #pragma unroll
            for (int mt = 0; mt < 4; ++mt)
                acc[mt][nt] = __builtin_amdgcn_wmma_f32_16x16x32_bf16(
                    false, a[mt].v, false, b.v, (short)0, acc[mt][nt], false, false);
        }
    }

    // Epilogue 1: bias + relu -> bf16 h in LDS
    #pragma unroll
    for (int nt = 0; nt < 4; ++nt) {
        int   N    = wave * 64 + nt * 16 + laneLo;
        float bias = b1[N];
        #pragma unroll
        for (int mt = 0; mt < 4; ++mt) {
            v8f c = acc[mt][nt];
            #pragma unroll
            for (int r = 0; r < 8; ++r) {
                int   M = mt * 16 + r + 8 * laneHi;
                float v = c[r] + bias;
                v = v > 0.f ? v : 0.f;
                hbuf[M * 512 + N] = f2bf(v);
            }
            acc[mt][nt] = zero8;   // recycle accumulators for phase 2
        }
    }
    __syncthreads();

    // ---------------- Phase 2: h @ W2 ----------------
    for (int kt = 0; kt < K2_PAD / 32; ++kt) {
        Frag a[4];
        #pragma unroll
        for (int mt = 0; mt < 4; ++mt) {
            const u16* p = &hbuf[(mt * 16 + laneLo) * 512 + kt * 32 + laneHi * 8];
            a[mt].h[0] = *(const u16x8*)p;
            a[mt].h[1] = *(const u16x8*)(p + 16);
        }
        #pragma unroll
        for (int nt = 0; nt < 4; ++nt) {
            int ntg = wave * 4 + nt;
            const u16* bp = &W2p[(((long)kt * 32 + ntg) * 32 + lane) * 16];
            Frag b;
            b.h[0] = *(const u16x8*)bp;
            b.h[1] = *(const u16x8*)(bp + 8);
            #pragma unroll
            for (int mt = 0; mt < 4; ++mt)
                acc[mt][nt] = __builtin_amdgcn_wmma_f32_16x16x32_bf16(
                    false, a[mt].v, false, b.v, (short)0, acc[mt][nt], false, false);
        }
    }

    // Epilogue 2: bias + relu, scatter-add into aggSum[row]
    #pragma unroll
    for (int nt = 0; nt < 4; ++nt) {
        int   N    = wave * 64 + nt * 16 + laneLo;
        float bias = b2[N];
        #pragma unroll
        for (int mt = 0; mt < 4; ++mt) {
            v8f c = acc[mt][nt];
            #pragma unroll
            for (int r = 0; r < 8; ++r) {
                int  M = mt * 16 + r + 8 * laneHi;
                long e = eBase + M;
                if (e < N_EDGES_C) {
                    float v = c[r] + bias;
                    v = v > 0.f ? v : 0.f;
                    atomicAdd(&aggSum[(long)rowL[M] * 512 + N], v);
                }
            }
        }
    }
}

// ---------------------------------------------------------------------------
// Node MLP: z = relu([x, agg, u[batch]] @ W3 + b3); out = z @ W4 + b4
// WG = 256 threads (8 waves), M = 64 nodes, each wave owns 64 columns.
// Final 512->1 layer fused via LDS float-atomic reduction.
// ---------------------------------------------------------------------------
__global__ __launch_bounds__(256) void node_mlp_kernel(
    const float* __restrict__ x, const int* __restrict__ batch,
    const float* __restrict__ u, const float* __restrict__ b3,
    const float* __restrict__ b4, const float* __restrict__ W4,
    const u16* __restrict__ W3p, const float* __restrict__ aggSum,
    const float* __restrict__ cnt, float* __restrict__ out) {

    __shared__ __align__(16) u16 atile[NODE_M * 32];      // 4 KB
    __shared__ float rcp[NODE_M];
    __shared__ int   bidx[NODE_M];
    __shared__ float accL[NODE_M];

    const int  t      = threadIdx.x;
    const int  wave   = t >> 5;
    const int  lane   = t & 31;
    const int  laneHi = lane >> 4;
    const int  laneLo = lane & 15;
    const long nBase  = (long)blockIdx.x * NODE_M;

    if (t < NODE_M) {
        long n  = nBase + t;
        float c = (n < N_NODES_C) ? cnt[n] : 1.0f;
        rcp[t]  = 1.0f / fmaxf(c, 1.0f);
        bidx[t] = (n < N_NODES_C) ? batch[n] : 0;
        accL[t] = 0.0f;
    }
    __syncthreads();

    const v8f zero8 = {0.f, 0.f, 0.f, 0.f, 0.f, 0.f, 0.f, 0.f};
    v8f acc[4][4];
    #pragma unroll
    for (int mt = 0; mt < 4; ++mt)
        #pragma unroll
        for (int nt = 0; nt < 4; ++nt) acc[mt][nt] = zero8;

    for (int kt = 0; kt < K3_PAD / 32; ++kt) {
        __syncthreads();
        #pragma unroll
        for (int q = 0; q < (NODE_M * 32) / 256; ++q) {
            int idx = q * 256 + t;
            int r = idx >> 5, c = idx & 31;
            int k = kt * 32 + c;
            long n = nBase + r;
            float v = 0.0f;
            if (n < N_NODES_C) {
                if (k < NODE_F_C)
                    v = x[n * NODE_F_C + k];
                else if (k < NODE_F_C + HID_C)
                    v = aggSum[n * (long)HID_C + (k - NODE_F_C)] * rcp[r];
                else if (k < K3_ORIG)
                    v = u[(long)bidx[r] * U_F_C + (k - NODE_F_C - HID_C)];
            }
            atile[idx] = f2bf(v);
        }
        __syncthreads();

        Frag a[4];
        #pragma unroll
        for (int mt = 0; mt < 4; ++mt) {
            const u16* p = &atile[(mt * 16 + laneLo) * 32 + laneHi * 8];
            a[mt].h[0] = *(const u16x8*)p;
            a[mt].h[1] = *(const u16x8*)(p + 16);
        }
        #pragma unroll
        for (int nt = 0; nt < 4; ++nt) {
            int ntg = wave * 4 + nt;
            const u16* bp = &W3p[(((long)kt * 32 + ntg) * 32 + lane) * 16];
            Frag b;
            b.h[0] = *(const u16x8*)bp;
            b.h[1] = *(const u16x8*)(bp + 8);
            #pragma unroll
            for (int mt = 0; mt < 4; ++mt)
                acc[mt][nt] = __builtin_amdgcn_wmma_f32_16x16x32_bf16(
                    false, a[mt].v, false, b.v, (short)0, acc[mt][nt], false, false);
        }
    }

    // Epilogue: relu(z)·W4 reduced per row through LDS atomics
    #pragma unroll
    for (int nt = 0; nt < 4; ++nt) {
        int   N    = wave * 64 + nt * 16 + laneLo;
        float bias = b3[N];
        float w4   = W4[N];
        #pragma unroll
        for (int mt = 0; mt < 4; ++mt) {
            v8f c = acc[mt][nt];
            #pragma unroll
            for (int r = 0; r < 8; ++r) {
                int   M = mt * 16 + r + 8 * laneHi;
                float v = c[r] + bias;
                v = v > 0.f ? v : 0.f;
                atomicAdd(&accL[M], v * w4);
            }
        }
    }
    __syncthreads();
    if (t < NODE_M) {
        long n = nBase + t;
        if (n < N_NODES_C) out[n] = accL[t] + b4[0];
    }
}

// ---------------------------------------------------------------------------
extern "C" void kernel_launch(void* const* d_in, const int* in_sizes, int n_in,
                              void* d_out, int out_size, void* d_ws, size_t ws_size,
                              hipStream_t stream) {
    const float* x     = (const float*)d_in[0];
    const int*   ei    = (const int*)d_in[1];
    const float* ea    = (const float*)d_in[2];
    const float* u     = (const float*)d_in[3];
    const int*   batch = (const int*)d_in[4];
    const float* W1    = (const float*)d_in[5];
    const float* b1    = (const float*)d_in[6];
    const float* W2    = (const float*)d_in[7];
    const float* b2    = (const float*)d_in[8];
    const float* W3    = (const float*)d_in[9];
    const float* b3    = (const float*)d_in[10];
    const float* W4    = (const float*)d_in[11];
    const float* b4    = (const float*)d_in[12];
    float* out = (float*)d_out;

    char* ws = (char*)d_ws;
    size_t off = 0;
    auto alloc = [&](size_t bytes) {
        char* p = ws + off;
        off += (bytes + 255) & ~(size_t)255;
        return p;
    };
    u16*   W1p    = (u16*)alloc((size_t)K1_PAD * 512 * 2);
    u16*   W2p    = (u16*)alloc((size_t)K2_PAD * 512 * 2);
    u16*   W3p    = (u16*)alloc((size_t)K3_PAD * 512 * 2);
    float* cnt    = (float*)alloc((size_t)N_NODES_C * 4);
    float* aggSum = (float*)alloc((size_t)N_NODES_C * HID_C * 4);

    hipMemsetAsync(cnt, 0, (size_t)N_NODES_C * 4, stream);
    hipMemsetAsync(aggSum, 0, (size_t)N_NODES_C * HID_C * 4, stream);

    pack_w_kernel<<<(K1_PAD * 512 + 255) / 256, 256, 0, stream>>>(W1, W1p, K1_ORIG, K1_PAD);
    pack_w_kernel<<<(K2_PAD * 512 + 255) / 256, 256, 0, stream>>>(W2, W2p, 512, K2_PAD);
    pack_w_kernel<<<(K3_PAD * 512 + 255) / 256, 256, 0, stream>>>(W3, W3p, K3_ORIG, K3_PAD);

    count_kernel<<<(N_EDGES_C + 255) / 256, 256, 0, stream>>>(ei, cnt);

    edge_mlp_kernel<<<(N_EDGES_C + EDGE_M - 1) / EDGE_M, 256, 0, stream>>>(
        x, ei, ea, b1, b2, W1p, W2p, aggSum);

    node_mlp_kernel<<<(N_NODES_C + NODE_M - 1) / NODE_M, 256, 0, stream>>>(
        x, batch, u, b3, b4, W4, W3p, aggSum, cnt, out);
}
